// PosteriorEncoder_214748365419
// MI455X (gfx1250) — compile-verified, matrix operand order checked by reading, not run on hardware
//
#include <hip/hip_runtime.h>
#include <hip/hip_bf16.h>

typedef __attribute__((ext_vector_type(16))) _Float16 v16h;
typedef __attribute__((ext_vector_type(8)))  _Float16 v8h;
typedef __attribute__((ext_vector_type(8)))  float    v8f;
typedef int vec4i __attribute__((vector_size(16)));   // matches async-LDS builtin param

// Problem constants (match reference)
constexpr int BATCH = 8;
constexpr int NPTS  = 8192;
constexpr int CF    = 4;
constexpr int CONDD = 128;
constexpr int KNN   = 16;
constexpr int SLOT  = KNN + 1;   // K neighbors + self loop
constexpr int NO1   = 2048, NO2 = 512, NO3 = 128;

// gfx1250 async global->LDS path (guarded: falls back to plain LDS copy)
#if defined(__HIP_DEVICE_COMPILE__) && defined(__gfx1250__)
#  if __has_builtin(__builtin_amdgcn_global_load_async_to_lds_b128)
#    define ASYNC_LDS 1
#  endif
#  if __has_builtin(__builtin_amdgcn_s_wait_asynccnt)
#    define WAIT_ASYNC_BUILTIN 1
#  endif
#endif

__device__ __forceinline__ void wait_async_all() {
#if defined(ASYNC_LDS)
#  if defined(WAIT_ASYNC_BUILTIN)
  __builtin_amdgcn_s_wait_asynccnt(0);
#  else
  asm volatile("s_wait_asynccnt 0x0" ::: "memory");
#  endif
#endif
}

// ---------------------------------------------------------------------------
// Farthest point sampling: one workgroup per sample, dmin in registers,
// wave32 shfl reduction + LDS cross-wave reduction. The 2048+512+128
// sequential argmax steps are the latency-critical chain of the whole net.
// ---------------------------------------------------------------------------
__global__ __launch_bounds__(1024) void fps_kernel(
    const float* __restrict__ pos, int n, int n_out, int* __restrict__ idx_out) {
  const int b = blockIdx.x;
  const float* p = pos + (size_t)b * n * 3;
  int* out = idx_out + (size_t)b * n_out;
  __shared__ float wval[32];
  __shared__ int   widx[32];
  __shared__ int   s_cur;
  const int tid = threadIdx.x;

  float dmin[8];
#pragma unroll
  for (int u = 0; u < 8; ++u) dmin[u] = 1e10f;

  if (tid == 0) { s_cur = 0; out[0] = 0; }
  __syncthreads();

  for (int it = 1; it < n_out; ++it) {
    const int cur = s_cur;
    const float lx = p[cur * 3 + 0], ly = p[cur * 3 + 1], lz = p[cur * 3 + 2];
    float bv = -1.0f;
    int   bi = 0;
#pragma unroll
    for (int u = 0; u < 8; ++u) {
      const int i = tid + u * 1024;
      if (i < n) {
        const float dx = p[i * 3 + 0] - lx;
        const float dy = p[i * 3 + 1] - ly;
        const float dz = p[i * 3 + 2] - lz;
        const float d  = dx * dx + dy * dy + dz * dz;
        const float dm = fminf(dmin[u], d);
        dmin[u] = dm;
        if (dm > bv) { bv = dm; bi = i; }
      }
    }
#pragma unroll
    for (int off = 16; off > 0; off >>= 1) {
      const float ov = __shfl_down(bv, off);
      const int   oi = __shfl_down(bi, off);
      if (ov > bv) { bv = ov; bi = oi; }
    }
    if ((tid & 31) == 0) { wval[tid >> 5] = bv; widx[tid >> 5] = bi; }
    __syncthreads();
    if (tid < 32) {
      float v = wval[tid];
      int   ix = widx[tid];
#pragma unroll
      for (int off = 16; off > 0; off >>= 1) {
        const float ov = __shfl_down(v, off);
        const int   oi = __shfl_down(ix, off);
        if (ov > v) { v = ov; ix = oi; }
      }
      if (tid == 0) { s_cur = ix; out[it] = ix; }
    }
    __syncthreads();
  }
}

// Gather sampled center positions: q[b,i,:] = pos[b, idx[b,i], :]
__global__ void gather_q_kernel(const float* __restrict__ pos,
                                const int* __restrict__ idx,
                                float* __restrict__ q,
                                int n_src, int n_out, int total) {
  const int t = blockIdx.x * blockDim.x + threadIdx.x;
  if (t >= total) return;
  const int b = t / (n_out * 3);
  const int r = t % (n_out * 3);
  const int i = r / 3, d = r % 3;
  q[t] = pos[((size_t)b * n_src + idx[(size_t)b * n_out + i]) * 3 + d];
}

// kNN: one thread per center, 16-deep insertion list (k smallest sq-dists).
__global__ void knn_kernel(const float* __restrict__ src,
                           const float* __restrict__ q,
                           int* __restrict__ nbr,
                           int n_src, int n_out, int total) {
  const int t = blockIdx.x * blockDim.x + threadIdx.x;
  if (t >= total) return;
  const int b = t / n_out, i = t % n_out;
  const float* qp = q + ((size_t)b * n_out + i) * 3;
  const float qx = qp[0], qy = qp[1], qz = qp[2];
  const float* s = src + (size_t)b * n_src * 3;
  float bd[KNN];
  int   bi[KNN];
#pragma unroll
  for (int k = 0; k < KNN; ++k) { bd[k] = 1e30f; bi[k] = 0; }
  for (int j = 0; j < n_src; ++j) {
    const float dx = s[j * 3 + 0] - qx;
    const float dy = s[j * 3 + 1] - qy;
    const float dz = s[j * 3 + 2] - qz;
    const float d  = dx * dx + dy * dy + dz * dz;
    if (d < bd[KNN - 1]) {
      int k = KNN - 1;
      while (k > 0 && bd[k - 1] > d) { bd[k] = bd[k - 1]; bi[k] = bi[k - 1]; --k; }
      bd[k] = d; bi[k] = j;
    }
  }
  int* o = nbr + ((size_t)b * n_out + i) * KNN;
#pragma unroll
  for (int k = 0; k < KNN; ++k) o[k] = bi[k];
}

// Build f16 feature rows: row = (b, center i, slot s) -> [x_j | pos_j - q_i | 0pad]
__global__ void gather_feat_kernel(const float* __restrict__ x, int dx,
                                   const float* __restrict__ pos,
                                   const float* __restrict__ q,
                                   const int* __restrict__ nbr,
                                   _Float16* __restrict__ A,
                                   int n_src, int n_out, int kpad, int total_rows) {
  const int r = blockIdx.x * blockDim.x + threadIdx.x;
  if (r >= total_rows) return;
  const int per = n_out * SLOT;
  const int b = r / per;
  const int rem = r % per;
  const int i = rem / SLOT;
  const int s = rem % SLOT;
  const int j = (s < KNN) ? nbr[((size_t)b * n_out + i) * KNN + s] : i;  // self loop
  _Float16* a = A + (size_t)r * kpad;
  const float* xs = x + ((size_t)b * n_src + j) * dx;
  int t = 0;
  for (; t < dx; ++t) a[t] = (_Float16)xs[t];
  const float* pj = pos + ((size_t)b * n_src + j) * 3;
  const float* qi = q + ((size_t)b * n_out + i) * 3;
  a[t++] = (_Float16)(pj[0] - qi[0]);
  a[t++] = (_Float16)(pj[1] - qi[1]);
  a[t++] = (_Float16)(pj[2] - qi[2]);
  for (; t < kpad; ++t) a[t] = (_Float16)0.0f;
}

// ---------------------------------------------------------------------------
// WMMA GEMM:  Out = act( A[M,KD] @ Wt^T + bias + S ),  Wt stored [N, KD].
// One wave per 16x16 tile, 4 waves/block stacked along M. The B (weight)
// tile is identical for all 4 waves, so it is staged once per block into LDS
// via the gfx1250 async global->LDS path (ASYNCcnt + s_wait_asynccnt), and
// fragments are then served by ds_load_b128. KD is a template parameter so
// the K-loop fully unrolls and the A-stream loads can be clause-batched
// instead of draining loadcnt to zero before every WMMA.
// Fragment layout per the 16-bit A/B spec: lane l holds row (l&15); lanes
// 0-15 carry K in [0,8)+[16,24), lanes 16-31 carry K in [8,16)+[24,32).
// ---------------------------------------------------------------------------
template <int KD>
__global__ __launch_bounds__(128) void wmma_gemm_kernel(
    const _Float16* __restrict__ A, const _Float16* __restrict__ Wt,
    const float* __restrict__ bias, const float* __restrict__ S,
    float* __restrict__ Out, _Float16* __restrict__ OutH,
    int M, int N, int relu) {
  __shared__ _Float16 sB[16 * KD];

  const int lane = threadIdx.x & 31;
  const int wave = threadIdx.x >> 5;
  const int n0 = blockIdx.x * 16;
  const int m0 = blockIdx.y * 64 + wave * 16;

  // Cooperative stage of the 16 x KD weight tile into LDS (16B per lane).
  {
    const _Float16* Bsrc = Wt + (size_t)n0 * KD;
    constexpr int CHUNKS = (16 * KD) / 8;  // 8 halves = 16 bytes per chunk
    for (int c = threadIdx.x; c < CHUNKS; c += 128) {
#if defined(ASYNC_LDS)
      __builtin_amdgcn_global_load_async_to_lds_b128(
          (__attribute__((address_space(1))) vec4i*)(Bsrc + (size_t)c * 8),
          (__attribute__((address_space(3))) vec4i*)(sB + (size_t)c * 8),
          0, 0);
#else
      *(v8h*)(sB + (size_t)c * 8) = *(const v8h*)(Bsrc + (size_t)c * 8);
#endif
    }
    wait_async_all();
  }
  __syncthreads();

  if (m0 >= M) return;

  const int khalf = (lane >> 4) * 8;
  const _Float16* arow = A  + (size_t)(m0 + (lane & 15)) * KD + khalf;
  const _Float16* brow = sB + (size_t)(lane & 15) * KD + khalf;

  v8f acc = {};
#pragma unroll
  for (int k0 = 0; k0 < KD; k0 += 32) {
    const v8h a0 = *(const v8h*)(arow + k0);
    const v8h a1 = *(const v8h*)(arow + k0 + 16);
    const v8h b0 = *(const v8h*)(brow + k0);
    const v8h b1 = *(const v8h*)(brow + k0 + 16);
    v16h af, bf;
#pragma unroll
    for (int t = 0; t < 8; ++t) {
      af[t] = a0[t]; af[t + 8] = a1[t];
      bf[t] = b0[t]; bf[t + 8] = b1[t];
    }
    acc = __builtin_amdgcn_wmma_f32_16x16x32_f16(
        false, af, false, bf, (short)0, acc, false, false);
  }

  const int nc = n0 + (lane & 15);
  const int mr = m0 + (lane >> 4) * 8;
  const float bv = bias ? bias[nc] : 0.0f;
#pragma unroll
  for (int j = 0; j < 8; ++j) {
    const int row = mr + j;
    float v = acc[j] + bv;
    if (S)    v += S[(size_t)row * N + nc];
    if (relu) v = fmaxf(v, 0.0f);
    if (Out)  Out[(size_t)row * N + nc] = v;
    if (OutH) OutH[(size_t)row * N + nc] = (_Float16)v;
  }
}

// Max over the 17 message slots of each center.
__global__ void aggregate_kernel(const float* __restrict__ O,
                                 float* __restrict__ xout,
                                 int dout, int total) {
  const int t = blockIdx.x * blockDim.x + threadIdx.x;
  if (t >= total) return;
  const int c  = t % dout;
  const int bi = t / dout;
  const float* o = O + (size_t)bi * SLOT * dout + c;
  float m = o[0];
#pragma unroll
  for (int s = 1; s < SLOT; ++s) m = fmaxf(m, o[(size_t)s * dout]);
  xout[t] = m;
}

// Global max pool over the 128 SA3 centers + concat cond -> f16 rows [16,384].
__global__ void build_final_kernel(const float* __restrict__ x3,
                                   const float* __restrict__ cond,
                                   _Float16* __restrict__ A) {
  const int b = blockIdx.x;   // 16 rows (8 real + 8 zero pad)
  const int t = threadIdx.x;  // 384 features
  _Float16* a = A + b * 384 + t;
  if (b >= BATCH) { *a = (_Float16)0.0f; return; }
  float v;
  if (t < 256) {
    const float* p = x3 + (size_t)b * NO3 * 256 + t;
    v = p[0];
    for (int i = 1; i < NO3; ++i) v = fmaxf(v, p[(size_t)i * 256]);
  } else {
    v = cond[b * CONDD + (t - 256)];
  }
  *a = (_Float16)v;
}

// Copy rows 0..7 of the two [16,64] head outputs into d_out (mu then log_var).
__global__ void copy_heads_kernel(const float* __restrict__ mu,
                                  const float* __restrict__ lv,
                                  float* __restrict__ out) {
  const int t = blockIdx.x * blockDim.x + threadIdx.x;
  if (t >= 1024) return;
  out[t] = (t < 512) ? mu[t] : lv[t - 512];
}

// Convert f32 weight [din,dout] -> f16 transposed+padded [dout,kpad].
__global__ void convert_weight_kernel(const float* __restrict__ W,
                                      _Float16* __restrict__ Wt,
                                      int din, int dout, int kpad) {
  const int t = blockIdx.x * blockDim.x + threadIdx.x;
  if (t >= dout * kpad) return;
  const int n = t / kpad, k = t % kpad;
  Wt[t] = (k < din) ? (_Float16)W[(size_t)k * dout + n] : (_Float16)0.0f;
}

// ---------------------------------------------------------------------------
extern "C" void kernel_launch(void* const* d_in, const int* in_sizes, int n_in,
                              void* d_out, int out_size, void* d_ws, size_t ws_size,
                              hipStream_t stream) {
  const float* y    = (const float*)d_in[0];
  const float* qpos = (const float*)d_in[1];
  const float* cond = (const float*)d_in[2];

  // params, insertion order: sa1{l1,l2,sc} sa2{..} sa3{..} final{..} mu lv; each {W,b}
  const float* W[14]; const float* Bv[14];
  {
    int pi = 3;
    for (int w = 0; w < 14; ++w) { W[w] = (const float*)d_in[pi++]; Bv[w] = (const float*)d_in[pi++]; }
  }
  static const int ldin[14]  = {7, 32, 7,   67, 64, 67,   131, 128, 131,  384, 512, 384, 512, 512};
  static const int ldout[14] = {32, 64, 64, 64, 128, 128, 128, 256, 256,  512, 512, 512, 64, 64};
  static const int lkpad[14] = {32, 32, 32, 96, 64, 96,   160, 128, 160,  384, 512, 384, 512, 512};

  // workspace layout
  char* base = (char*)d_ws;
  size_t off = 0;
  auto alloc = [&](size_t bytes) -> void* {
    void* p = base + off;
    off = (off + bytes + 255) & ~(size_t)255;
    return p;
  };

  _Float16* Wt[14];
  for (int w = 0; w < 14; ++w) Wt[w] = (_Float16*)alloc((size_t)ldout[w] * lkpad[w] * 2);

  int*   idx1 = (int*)alloc((size_t)BATCH * NO1 * 4);
  int*   idx2 = (int*)alloc((size_t)BATCH * NO2 * 4);
  int*   idx3 = (int*)alloc((size_t)BATCH * NO3 * 4);
  float* q1   = (float*)alloc((size_t)BATCH * NO1 * 3 * 4);
  float* q2   = (float*)alloc((size_t)BATCH * NO2 * 3 * 4);
  float* q3   = (float*)alloc((size_t)BATCH * NO3 * 3 * 4);
  int*   nbr1 = (int*)alloc((size_t)BATCH * NO1 * KNN * 4);
  int*   nbr2 = (int*)alloc((size_t)BATCH * NO2 * KNN * 4);
  int*   nbr3 = (int*)alloc((size_t)BATCH * NO3 * KNN * 4);
  float* x2   = (float*)alloc((size_t)BATCH * NO1 * 64 * 4);
  float* x3   = (float*)alloc((size_t)BATCH * NO2 * 128 * 4);
  float* x4   = (float*)alloc((size_t)BATCH * NO3 * 256 * 4);

  const size_t R1 = (size_t)BATCH * NO1 * SLOT;   // 278528 rows (multiple of 64)
  _Float16* Abuf = (_Float16*)alloc(R1 * 32 * 2);   // max per-level A (17.8 MB)
  _Float16* Hbuf = (_Float16*)alloc(R1 * 32 * 2);   // max hidden f16
  float*    Sbuf = (float*)alloc(R1 * 64 * 4);      // max shortcut/out f32

  _Float16* Afin  = (_Float16*)alloc((size_t)16 * 384 * 2);
  _Float16* Hfin  = (_Float16*)alloc((size_t)16 * 512 * 2);
  _Float16* HfinB = (_Float16*)alloc((size_t)16 * 512 * 2);
  float*    Sfin  = (float*)alloc((size_t)16 * 512 * 4);
  float*    muO   = (float*)alloc((size_t)16 * 64 * 4);
  float*    lvO   = (float*)alloc((size_t)16 * 64 * 4);

  // weight conversion (small, once per launch, deterministic)
  for (int w = 0; w < 14; ++w) {
    const int total = ldout[w] * lkpad[w];
    convert_weight_kernel<<<(total + 255) / 256, 256, 0, stream>>>(
        W[w], Wt[w], ldin[w], ldout[w], lkpad[w]);
  }

  auto gemm = [&](const _Float16* A, const _Float16* Wtt, const float* bias,
                  const float* S, float* Out, _Float16* OutH,
                  int M, int N, int Kd, int relu) {
    dim3 grid(N / 16, (M + 63) / 64);
    switch (Kd) {
      case 32:  wmma_gemm_kernel<32> <<<grid, 128, 0, stream>>>(A, Wtt, bias, S, Out, OutH, M, N, relu); break;
      case 64:  wmma_gemm_kernel<64> <<<grid, 128, 0, stream>>>(A, Wtt, bias, S, Out, OutH, M, N, relu); break;
      case 96:  wmma_gemm_kernel<96> <<<grid, 128, 0, stream>>>(A, Wtt, bias, S, Out, OutH, M, N, relu); break;
      case 128: wmma_gemm_kernel<128><<<grid, 128, 0, stream>>>(A, Wtt, bias, S, Out, OutH, M, N, relu); break;
      case 160: wmma_gemm_kernel<160><<<grid, 128, 0, stream>>>(A, Wtt, bias, S, Out, OutH, M, N, relu); break;
      case 384: wmma_gemm_kernel<384><<<grid, 128, 0, stream>>>(A, Wtt, bias, S, Out, OutH, M, N, relu); break;
      case 512: wmma_gemm_kernel<512><<<grid, 128, 0, stream>>>(A, Wtt, bias, S, Out, OutH, M, N, relu); break;
      default: break;
    }
  };

  struct Level {
    const float* src_pos; int n_src;
    const float* x; int dx;
    int n_out; int kpad; int dh; int dout;
    int w0;                       // index of l1 weights (l2=w0+1, sc=w0+2)
    int* idx; float* q; int* nbr; float* xout;
  };
  Level lv[3] = {
    {qpos, NPTS, y,  CF,  NO1, 32,  32,  64,  0, idx1, q1, nbr1, x2},
    {q1,   NO1,  x2, 64,  NO2, 96,  64,  128, 3, idx2, q2, nbr2, x3},
    {q2,   NO2,  x3, 128, NO3, 160, 128, 256, 6, idx3, q3, nbr3, x4},
  };

  for (int L = 0; L < 3; ++L) {
    const Level& v = lv[L];
    fps_kernel<<<BATCH, 1024, 0, stream>>>(v.src_pos, v.n_src, v.n_out, v.idx);
    {
      const int total = BATCH * v.n_out * 3;
      gather_q_kernel<<<(total + 255) / 256, 256, 0, stream>>>(
          v.src_pos, v.idx, v.q, v.n_src, v.n_out, total);
    }
    {
      const int total = BATCH * v.n_out;
      knn_kernel<<<(total + 255) / 256, 256, 0, stream>>>(
          v.src_pos, v.q, v.nbr, v.n_src, v.n_out, total);
    }
    const int R = BATCH * v.n_out * SLOT;
    gather_feat_kernel<<<(R + 255) / 256, 256, 0, stream>>>(
        v.x, v.dx, v.src_pos, v.q, v.nbr, Abuf, v.n_src, v.n_out, v.kpad, R);
    // shortcut: S = A @ Wsc + bsc
    gemm(Abuf, Wt[v.w0 + 2], Bv[v.w0 + 2], nullptr, Sbuf, nullptr, R, v.dout, v.kpad, 0);
    // hidden: H = relu(A @ W1 + b1) (f16 for next GEMM)
    gemm(Abuf, Wt[v.w0 + 0], Bv[v.w0 + 0], nullptr, nullptr, Hbuf, R, v.dh, v.kpad, 1);
    // out: relu(H @ W2 + b2 + S), in place over Sbuf
    gemm(Hbuf, Wt[v.w0 + 1], Bv[v.w0 + 1], Sbuf, Sbuf, nullptr, R, v.dout, v.dh, 1);
    {
      const int total = BATCH * v.n_out * v.dout;
      aggregate_kernel<<<(total + 255) / 256, 256, 0, stream>>>(Sbuf, v.xout, v.dout, total);
    }
  }

  // final: global max pool + cond concat -> ResMLP(384->512->512) -> heads
  build_final_kernel<<<16, 384, 0, stream>>>(x4, cond, Afin);
  gemm(Afin,  Wt[11], Bv[11], nullptr, Sfin, nullptr, 16, 512, 384, 0);          // shortcut
  gemm(Afin,  Wt[9],  Bv[9],  nullptr, nullptr, Hfin, 16, 512, 384, 1);          // hidden
  gemm(Hfin,  Wt[10], Bv[10], Sfin, Sfin, HfinB, 16, 512, 512, 1);               // h = relu(...)
  gemm(HfinB, Wt[12], Bv[12], nullptr, muO, nullptr, 16, 64, 512, 0);            // mu
  gemm(HfinB, Wt[13], Bv[13], nullptr, lvO, nullptr, 16, 64, 512, 0);            // log_var
  copy_heads_kernel<<<4, 256, 0, stream>>>(muO, lvO, (float*)d_out);

  (void)in_sizes; (void)n_in; (void)out_size; (void)ws_size;
}
